// ShamiLayer_90237262889111
// MI455X (gfx1250) — matrix operand order repaired and checked by reading, not compiled, not run on hardware
//
#include <hip/hip_runtime.h>
#include <cstdint>
#include <cstddef>

// ---------------------------------------------------------------------------
// Transformer layer (RMSNorm -> QKV -> RoPE -> causal flash attention -> WO
// -> residual -> RMSNorm -> SwiGLU FFN -> residual) for MI455X (gfx1250).
// All GEMM-shaped math runs on v_wmma_f32_16x16x32_bf16 (bf16 in, f32 acc).
// GEMM tiles are staged global->LDS via GLOBAL_LOAD_ASYNC_TO_LDS_B128 when
// the toolchain exposes the builtin (ASYNCcnt-tracked CDNA5 path).
// ---------------------------------------------------------------------------

typedef __attribute__((ext_vector_type(16))) __bf16 bf16x16;
typedef __attribute__((ext_vector_type(8)))  __bf16 bf16x8;
typedef __attribute__((ext_vector_type(8)))  float  f32x8;
typedef int v4i __attribute__((vector_size(16)));   // matches builtin's V4i

#define CB  2
#define CS  2048
#define CD  2048
#define CH  16
#define CDH 128
#define CFF 5632
#define CM  (CB * CS)      // 4096 rows
#define CBH (CB * CH)      // 32 (b,h) pairs

#if __has_builtin(__builtin_amdgcn_global_load_async_to_lds_b128)
#define HAVE_ASYNC_LDS 1
#else
#define HAVE_ASYNC_LDS 0
#endif

__device__ __forceinline__ __bf16 f2bf(float f) {
  unsigned u = __builtin_bit_cast(unsigned, f);
  unsigned r = u + 0x7fffu + ((u >> 16) & 1u);   // round-to-nearest-even
  unsigned short h = (unsigned short)(r >> 16);
  return __builtin_bit_cast(__bf16, h);
}
__device__ __forceinline__ float bf2f(__bf16 b) {
  unsigned short h = __builtin_bit_cast(unsigned short, b);
  unsigned u = ((unsigned)h) << 16;
  return __builtin_bit_cast(float, u);
}
__device__ __forceinline__ bf16x16 cat8(bf16x8 lo, bf16x8 hi) {
  return __builtin_shufflevector(lo, hi, 0,1,2,3,4,5,6,7,8,9,10,11,12,13,14,15);
}
__device__ __forceinline__ f32x8 wmma_bf16(bf16x16 a, bf16x16 b, f32x8 c) {
  // 8-arg form: (neg_a, A, neg_b, B, c_mod, C, reuse_a, reuse_b)
  return __builtin_amdgcn_wmma_f32_16x16x32_bf16(false, a, false, b, (short)0, c,
                                                 false, false);
}

// 16B global -> LDS copy: async DMA path on gfx1250, VGPR round-trip fallback.
__device__ __forceinline__ void cp16_g2l(const __bf16* g, __bf16* l) {
#if HAVE_ASYNC_LDS
  __builtin_amdgcn_global_load_async_to_lds_b128(
      (__attribute__((address_space(1))) v4i*)(g),
      (__attribute__((address_space(3))) v4i*)(l), 0, 0);
#else
  *(bf16x8*)l = *(const bf16x8*)g;
#endif
}
__device__ __forceinline__ void wait_async0() {
#if HAVE_ASYNC_LDS
#if __has_builtin(__builtin_amdgcn_s_wait_asynccnt)
  __builtin_amdgcn_s_wait_asynccnt(0);
#else
  asm volatile("s_wait_asynccnt 0x0" ::: "memory");
#endif
#endif
}

// ---------------------------------------------------------------------------
// Weight transpose + fp32->bf16 convert:  wt[n*K + k] = bf16(w[k*N + n])
// Gives the GEMM a K-contiguous B matrix so B-fragments are 16B vector loads.
// ---------------------------------------------------------------------------
__global__ __launch_bounds__(256) void wtrans_kernel(const float* __restrict__ w,
                                                     __bf16* __restrict__ wt,
                                                     long K, long N) {
  long t = (long)blockIdx.x * 256 + threadIdx.x;
  long total = K * N / 8;
  if (t >= total) return;
  long k8 = (t % (K / 8)) * 8;
  long n  = t / (K / 8);
  bf16x8 o;
  #pragma unroll
  for (int j = 0; j < 8; ++j) o[j] = f2bf(w[(k8 + j) * N + n]);
  *(bf16x8*)(wt + n * K + k8) = o;
}

// ---------------------------------------------------------------------------
// RMSNorm: f32 in -> bf16 out (one block per row of D=2048)
// ---------------------------------------------------------------------------
__global__ __launch_bounds__(256) void rmsnorm_kernel(const float* __restrict__ x,
                                                      const float* __restrict__ w,
                                                      __bf16* __restrict__ out) {
  const long row = blockIdx.x;
  const float* xr = x + row * (long)CD;
  float ss = 0.f;
  for (int i = threadIdx.x; i < CD; i += 256) { float v = xr[i]; ss += v * v; }
  #pragma unroll
  for (int m = 16; m; m >>= 1) ss += __shfl_xor(ss, m, 32);
  __shared__ float red[8];
  if ((threadIdx.x & 31) == 0) red[threadIdx.x >> 5] = ss;
  __syncthreads();
  float tot = 0.f;
  #pragma unroll
  for (int j = 0; j < 8; ++j) tot += red[j];
  float scale = rsqrtf(tot / (float)CD + 1e-6f);
  for (int i = threadIdx.x; i < CD; i += 256)
    out[row * (long)CD + i] = f2bf(xr[i] * scale * w[i]);
}

// ---------------------------------------------------------------------------
// bf16 GEMM: C[M,N] = A[M,K] x B[K,N], B provided transposed as Bt[N,K].
// Block tile 128x256x64, 8 waves in a 2x4 grid, each wave 64x64 via 4x4
// 16x16 WMMA accumulators -> 16 WMMA per 32-deep k-step per wave.
// EPI: 0=f32 store, 1=bf16 store, 2=f32 store of (c+aux), 3=bf16(silu(aux)*c)
// ---------------------------------------------------------------------------
#define GBM 128
#define GBN 256
#define GBK 64
#define GLDS 80   // padded element stride (160B rows -> 16B-aligned b128 ops)

template <int EPI>
__global__ __launch_bounds__(256, 1) void gemm_bf16_kernel(
    const __bf16* __restrict__ A, const __bf16* __restrict__ Bt,
    void* __restrict__ Cout, const float* __restrict__ aux,
    int M, int N, int K) {
  __shared__ __bf16 sA[GBM * GLDS];   // 20 KB
  __shared__ __bf16 sB[GBN * GLDS];   // 40 KB
  const int tid = threadIdx.x;
  const int wave = tid >> 5;
  const int lane = tid & 31;
  const int r = lane & 15;
  const int g = lane >> 4;
  const int wm = (wave >> 2) * 64;   // 2 waves along M
  const int wn = (wave & 3) * 64;    // 4 waves along N
  const long m0 = (long)blockIdx.y * GBM;
  const long n0 = (long)blockIdx.x * GBN;

  f32x8 acc[4][4];
  #pragma unroll
  for (int mt = 0; mt < 4; ++mt)
    #pragma unroll
    for (int nt = 0; nt < 4; ++nt)
      #pragma unroll
      for (int i = 0; i < 8; ++i) acc[mt][nt][i] = 0.f;

  for (int k0 = 0; k0 < K; k0 += GBK) {
    // cooperative tile stage: A 128x64, B 256x64 (bf16), 16B per op
    #pragma unroll
    for (int i = 0; i < 4; ++i) {      // A: 1024 chunks / 256 threads
      int chunk = i * 256 + tid;
      int row = chunk >> 3, cc = chunk & 7;
      cp16_g2l(A + (m0 + row) * (long)K + k0 + cc * 8, &sA[row * GLDS + cc * 8]);
    }
    #pragma unroll
    for (int i = 0; i < 8; ++i) {      // B: 2048 chunks / 256 threads
      int chunk = i * 256 + tid;
      int row = chunk >> 3, cc = chunk & 7;
      cp16_g2l(Bt + (n0 + row) * (long)K + k0 + cc * 8, &sB[row * GLDS + cc * 8]);
    }
    if (k0 + GBK < K) {  // gfx1250 global_prefetch_b8 of next tile
      __builtin_prefetch(A + (m0 + (tid >> 1)) * (long)K + k0 + GBK, 0, 3);
      __builtin_prefetch(Bt + (n0 + tid) * (long)K + k0 + GBK, 0, 3);
    }
    wait_async0();
    __syncthreads();
    #pragma unroll
    for (int kc = 0; kc < GBK; kc += 32) {
      bf16x16 af[4], bfr[4];
      #pragma unroll
      for (int mt = 0; mt < 4; ++mt) {
        // A frag: lane row = r, K runs {kc+8g..+7} and {kc+16+8g..+7}
        const __bf16* p = &sA[(wm + mt * 16 + r) * GLDS + kc + 8 * g];
        af[mt] = cat8(*(const bf16x8*)p, *(const bf16x8*)(p + 16));
      }
      #pragma unroll
      for (int nt = 0; nt < 4; ++nt) {
        // B frag: lane col = r, K runs {kc+8g..+7} and {kc+16+8g..+7}
        const __bf16* p = &sB[(wn + nt * 16 + r) * GLDS + kc + 8 * g];
        bfr[nt] = cat8(*(const bf16x8*)p, *(const bf16x8*)(p + 16));
      }
      #pragma unroll
      for (int mt = 0; mt < 4; ++mt)
        #pragma unroll
        for (int nt = 0; nt < 4; ++nt)
          acc[mt][nt] = wmma_bf16(af[mt], bfr[nt], acc[mt][nt]);
    }
    __syncthreads();
  }

  #pragma unroll
  for (int mt = 0; mt < 4; ++mt)
    #pragma unroll
    for (int nt = 0; nt < 4; ++nt)
      #pragma unroll
      for (int i = 0; i < 8; ++i) {
        long row = m0 + wm + mt * 16 + i + 8 * g;  // C layout: VGPR i -> M=i+8g
        long col = n0 + wn + nt * 16 + r;          //           N = lane%16
        long idx = row * (long)N + col;
        float v = acc[mt][nt][i];
        if constexpr (EPI == 0) {
          ((float*)Cout)[idx] = v;
        } else if constexpr (EPI == 1) {
          ((__bf16*)Cout)[idx] = f2bf(v);
        } else if constexpr (EPI == 2) {
          ((float*)Cout)[idx] = v + aux[idx];
        } else {
          float a = aux[idx];
          float s = a / (1.f + __expf(-a));        // silu
          ((__bf16*)Cout)[idx] = f2bf(s * v);
        }
      }
}

// ---------------------------------------------------------------------------
// RoPE + head-major transpose: q,k bf16 [B,S,H*DH] -> qh,kh bf16 [B*H,S,DH]
// ---------------------------------------------------------------------------
__global__ __launch_bounds__(256) void rope_kernel(const __bf16* __restrict__ q,
                                                   const __bf16* __restrict__ k,
                                                   const float* __restrict__ fcos,
                                                   const float* __restrict__ fsin,
                                                   __bf16* __restrict__ qh,
                                                   __bf16* __restrict__ kh) {
  int idx = blockIdx.x * 4 + (threadIdx.x >> 6);  // (b*S+s)*H + h
  int p = threadIdx.x & 63;                       // pair index 0..63
  int h = idx & (CH - 1);
  long bs = idx >> 4;
  int s = (int)(bs & (CS - 1));
  int b = (int)(bs >> 11);
  long in_off = bs * (long)(CH * CDH) + h * CDH + 2 * p;
  float c  = fcos[s * 64 + p];
  float sn = fsin[s * 64 + p];
  long out_off = (((long)(b * CH + h)) * CS + s) * CDH + 2 * p;

  float xr = bf2f(q[in_off]), xi = bf2f(q[in_off + 1]);
  qh[out_off]     = f2bf(xr * c - xi * sn);
  qh[out_off + 1] = f2bf(xr * sn + xi * c);
  xr = bf2f(k[in_off]); xi = bf2f(k[in_off + 1]);
  kh[out_off]     = f2bf(xr * c - xi * sn);
  kh[out_off + 1] = f2bf(xr * sn + xi * c);
}

// V -> Vt bf16 [B*H, DH, S]  (K-contiguous for the PV B-fragments)
__global__ __launch_bounds__(128) void vtrans_kernel(const __bf16* __restrict__ v,
                                                     __bf16* __restrict__ vhT) {
  int idx = blockIdx.x;                // (b*S+s)*H + h
  int h = idx & (CH - 1);
  long bs = idx >> 4;
  int s = (int)(bs & (CS - 1));
  int b = (int)(bs >> 11);
  int d = threadIdx.x;
  vhT[(((long)(b * CH + h)) * CDH + d) * CS + s] =
      v[bs * (long)(CH * CDH) + h * CDH + d];
}

// ---------------------------------------------------------------------------
// Causal flash attention, WMMA throughout.
// Grid: (S/64, B*H); block = 128 threads = 4 waves; wave owns 16 query rows.
// Per 64-wide KV chunk: 16 WMMA (scores) + 16 WMMA (PV).
// ---------------------------------------------------------------------------
__global__ __launch_bounds__(128, 1) void flash_attn_kernel(
    const __bf16* __restrict__ Qh,   // [BH, S, DH]
    const __bf16* __restrict__ Kh,   // [BH, S, DH]
    const __bf16* __restrict__ VhT,  // [BH, DH, S]
    __bf16* __restrict__ Out)        // [B, S, H*DH]
{
  const int bh = blockIdx.y;
  const int b = bh >> 4, h = bh & 15;
  const int qblk = blockIdx.x;
  const int wave = threadIdx.x >> 5;
  const int lane = threadIdx.x & 31;
  const int r = lane & 15, g = lane >> 4;
  const int qb = qblk * 64 + wave * 16;
  const float scale = 0.08838834764831845f;  // 1/sqrt(128)

  __shared__ __bf16 sP[4][16 * 80];          // per-wave P staging (padded)

  // Q fragments (A-layout): loaded once, reused over all KV chunks
  const __bf16* Qp = Qh + ((long)bh * CS + qb) * CDH;
  bf16x16 qf[4];
  #pragma unroll
  for (int kc = 0; kc < 4; ++kc) {
    const __bf16* p = Qp + (long)r * CDH + kc * 32 + 8 * g;
    qf[kc] = cat8(*(const bf16x8*)p, *(const bf16x8*)(p + 16));
  }

  f32x8 o[8];
  #pragma unroll
  for (int t = 0; t < 8; ++t)
    #pragma unroll
    for (int i = 0; i < 8; ++i) o[t][i] = 0.f;
  float mrow[8], lrow[8];
  #pragma unroll
  for (int i = 0; i < 8; ++i) { mrow[i] = -1e30f; lrow[i] = 0.f; }

  const int kv_end = qblk * 64 + 64;
  for (int kv0 = 0; kv0 < kv_end; kv0 += 64) {
    // ---- scores S = Q K^T (4 col-tiles x 4 K-steps) ----
    f32x8 s[4];
    #pragma unroll
    for (int tc = 0; tc < 4; ++tc) {
      f32x8 acc;
      #pragma unroll
      for (int i = 0; i < 8; ++i) acc[i] = 0.f;
      const __bf16* Kp = Kh + ((long)bh * CS + kv0 + tc * 16 + r) * CDH;
      #pragma unroll
      for (int kc = 0; kc < 4; ++kc) {
        const __bf16* p = Kp + kc * 32 + 8 * g;   // B frag: K runs 8g..+7, 16+8g..+7
        bf16x16 bfr = cat8(*(const bf16x8*)p, *(const bf16x8*)(p + 16));
        acc = wmma_bf16(qf[kc], bfr, acc);
      }
      s[tc] = acc;
    }
    // ---- online softmax (rows live in lanes of same g; reduce over r) ----
    #pragma unroll
    for (int i = 0; i < 8; ++i) {
      int qrow = qb + i + 8 * g;
      float mx = -1e30f;
      #pragma unroll
      for (int tc = 0; tc < 4; ++tc) {
        int kvcol = kv0 + tc * 16 + r;
        float v = s[tc][i] * scale;
        v = (kvcol > qrow) ? -1e30f : v;  // causal mask
        s[tc][i] = v;
        mx = fmaxf(mx, v);
      }
      #pragma unroll
      for (int msk = 8; msk; msk >>= 1) mx = fmaxf(mx, __shfl_xor(mx, msk, 32));
      float mnew = fmaxf(mrow[i], mx);
      float alpha = __expf(mrow[i] - mnew);
      mrow[i] = mnew;
      lrow[i] *= alpha;
      #pragma unroll
      for (int t = 0; t < 8; ++t) o[t][i] *= alpha;
      float rs = 0.f;
      #pragma unroll
      for (int tc = 0; tc < 4; ++tc) {
        float pv = __expf(s[tc][i] - mnew);
        s[tc][i] = pv;
        rs += pv;
      }
      #pragma unroll
      for (int msk = 8; msk; msk >>= 1) rs += __shfl_xor(rs, msk, 32);
      lrow[i] += rs;
    }
    // ---- P (C-layout) -> LDS -> A-layout fragments ----
    __bf16* Pw = &sP[wave][0];
    #pragma unroll
    for (int tc = 0; tc < 4; ++tc)
      #pragma unroll
      for (int i = 0; i < 8; ++i)
        Pw[(i + 8 * g) * 80 + tc * 16 + r] = f2bf(s[tc][i]);
    asm volatile("s_wait_dscnt 0x0" ::: "memory");  // same-wave LDS RAW fence
    bf16x16 pf[2];
    #pragma unroll
    for (int kc2 = 0; kc2 < 2; ++kc2) {
      const __bf16* p = &Pw[r * 80 + kc2 * 32 + 8 * g];
      pf[kc2] = cat8(*(const bf16x8*)p, *(const bf16x8*)(p + 16));
    }
    // ---- O += P V  (8 d-tiles x 2 K-steps) ----
    #pragma unroll
    for (int t = 0; t < 8; ++t) {
      const __bf16* Vp = VhT + (((long)bh) * CDH + t * 16 + r) * CS + kv0;
      f32x8 acc = o[t];
      #pragma unroll
      for (int kc2 = 0; kc2 < 2; ++kc2) {
        const __bf16* p = Vp + kc2 * 32 + 8 * g;
        bf16x16 bfr = cat8(*(const bf16x8*)p, *(const bf16x8*)(p + 16));
        acc = wmma_bf16(pf[kc2], bfr, acc);
      }
      o[t] = acc;
    }
  }
  // ---- normalize and store to [B, S, H*DH] bf16 ----
  #pragma unroll
  for (int t = 0; t < 8; ++t)
    #pragma unroll
    for (int i = 0; i < 8; ++i) {
      int qrow = qb + i + 8 * g;
      float v = o[t][i] / lrow[i];
      Out[((long)(b * CS + qrow)) * (CH * CDH) + h * CDH + t * 16 + r] = f2bf(v);
    }
}

// ---------------------------------------------------------------------------
// Host launcher
// ---------------------------------------------------------------------------
extern "C" void kernel_launch(void* const* d_in, const int* in_sizes, int n_in,
                              void* d_out, int out_size, void* d_ws, size_t ws_size,
                              hipStream_t stream) {
  (void)in_sizes; (void)n_in; (void)out_size; (void)ws_size;
  const float* hidden = (const float*)d_in[0];
  const float* fcos   = (const float*)d_in[1];
  const float* fsin   = (const float*)d_in[2];
  const float* wq     = (const float*)d_in[3];
  const float* wk     = (const float*)d_in[4];
  const float* wv     = (const float*)d_in[5];
  const float* wo     = (const float*)d_in[6];
  const float* w1     = (const float*)d_in[7];
  const float* w2     = (const float*)d_in[8];
  const float* w3     = (const float*)d_in[9];
  const float* anw    = (const float*)d_in[10];
  const float* fnw    = (const float*)d_in[11];
  float* out = (float*)d_out;

  const long D = CD, FF = CFF, M = CM;
  char* ws = (char*)d_ws;
  size_t off = 0;
  auto take = [&](size_t bytes) -> char* {
    char* p = ws + off;
    off += (bytes + 255) & ~(size_t)255;
    return p;
  };
  // bf16-transposed weights
  __bf16* wqT = (__bf16*)take(D * D * 2);
  __bf16* wkT = (__bf16*)take(D * D * 2);
  __bf16* wvT = (__bf16*)take(D * D * 2);
  __bf16* woT = (__bf16*)take(D * D * 2);
  __bf16* w1T = (__bf16*)take(D * FF * 2);
  __bf16* w2T = (__bf16*)take(FF * D * 2);
  __bf16* w3T = (__bf16*)take(D * FF * 2);
  // activations
  __bf16* xnorm = (__bf16*)take(M * D * 2);
  __bf16* qbuf  = (__bf16*)take(M * D * 2);
  __bf16* kbuf  = (__bf16*)take(M * D * 2);
  __bf16* vbuf  = (__bf16*)take(M * D * 2);
  __bf16* qh    = (__bf16*)take(M * D * 2);
  __bf16* kh    = (__bf16*)take(M * D * 2);
  __bf16* vhT   = (__bf16*)take(M * D * 2);
  __bf16* attn  = (__bf16*)take(M * D * 2);
  float*  hbuf  = (float*)take(M * D * 4);
  __bf16* ybuf  = (__bf16*)take(M * D * 2);
  __bf16* ffh   = (__bf16*)take(M * FF * 2);
  // g1 (f32, M*FF = 92.3MB) aliases the dead qbuf..vhT region (100.7MB):
  // q/k/v/qh/kh/vhT are consumed before the FFN GEMMs run.
  float* g1 = (float*)qbuf;

  auto wtb = [](long K, long N) { return (unsigned)((K * N / 8 + 255) / 256); };

  // 1) weight convert+transpose
  wtrans_kernel<<<wtb(D, D), 256, 0, stream>>>(wq, wqT, D, D);
  wtrans_kernel<<<wtb(D, D), 256, 0, stream>>>(wk, wkT, D, D);
  wtrans_kernel<<<wtb(D, D), 256, 0, stream>>>(wv, wvT, D, D);
  wtrans_kernel<<<wtb(D, D), 256, 0, stream>>>(wo, woT, D, D);
  wtrans_kernel<<<wtb(D, FF), 256, 0, stream>>>(w1, w1T, D, FF);
  wtrans_kernel<<<wtb(FF, D), 256, 0, stream>>>(w2, w2T, FF, D);
  wtrans_kernel<<<wtb(D, FF), 256, 0, stream>>>(w3, w3T, D, FF);

  // 2) attention-input RMSNorm
  rmsnorm_kernel<<<(unsigned)M, 256, 0, stream>>>(hidden, anw, xnorm);

  // 3) QKV projections (bf16 WMMA, bf16 out); block tile 128x256
  dim3 gDD((unsigned)(D / GBN), (unsigned)(M / GBM));    // (8, 32)
  gemm_bf16_kernel<1><<<gDD, 256, 0, stream>>>(xnorm, wqT, qbuf, nullptr, M, D, D);
  gemm_bf16_kernel<1><<<gDD, 256, 0, stream>>>(xnorm, wkT, kbuf, nullptr, M, D, D);
  gemm_bf16_kernel<1><<<gDD, 256, 0, stream>>>(xnorm, wvT, vbuf, nullptr, M, D, D);

  // 4) RoPE + head-major layout
  rope_kernel<<<(unsigned)(CB * CS * CH / 4), 256, 0, stream>>>(qbuf, kbuf, fcos,
                                                                fsin, qh, kh);
  vtrans_kernel<<<(unsigned)(CB * CS * CH), 128, 0, stream>>>(vbuf, vhT);

  // 5) causal flash attention
  dim3 gFA((unsigned)(CS / 64), (unsigned)CBH);
  flash_attn_kernel<<<gFA, 128, 0, stream>>>(qh, kh, vhT, attn);

  // 6) WO projection + residual -> h (f32)
  gemm_bf16_kernel<2><<<gDD, 256, 0, stream>>>(attn, woT, hbuf, hidden, M, D, D);

  // 7) FFN RMSNorm
  rmsnorm_kernel<<<(unsigned)M, 256, 0, stream>>>(hbuf, fnw, ybuf);

  // 8) SwiGLU FFN: g1 = y@w1 (f32); ffh = silu(g1) * (y@w3) (bf16)
  dim3 gDF((unsigned)(FF / GBN), (unsigned)(M / GBM));   // (22, 32)
  gemm_bf16_kernel<0><<<gDF, 256, 0, stream>>>(ybuf, w1T, g1, nullptr, M, FF, D);
  gemm_bf16_kernel<3><<<gDF, 256, 0, stream>>>(ybuf, w3T, ffh, g1, M, FF, D);

  // 9) down-projection + residual -> out (f32)
  gemm_bf16_kernel<2><<<gDD, 256, 0, stream>>>(ffh, w2T, out, hbuf, M, D, FF);
}
